// MambaBlock_43937515438319
// MI455X (gfx1250) — matrix operand order repaired and checked by reading, not compiled, not run on hardware
//
#include <hip/hip_runtime.h>

// Mamba block: B=2, L=1024, DIM=512, D=1024, N=16, R=32, K=4
#define BB    2
#define LL    1024
#define DIMM  512
#define DD    1024
#define NSTATE 16
#define RR    32
#define KCONV 4
#define ROWS  (BB * LL)   // 2048

typedef __attribute__((ext_vector_type(16))) __bf16 v16bf;
typedef __attribute__((ext_vector_type(8)))  __bf16 v8bf;
typedef __attribute__((ext_vector_type(8)))  float  v8f;

// ---------------------------------------------------------------------------
// f32 -> bf16 staging (row-major, used for A matrices)
// ---------------------------------------------------------------------------
__global__ void __launch_bounds__(256)
cvt_f32_to_bf16(const float* __restrict__ src, __bf16* __restrict__ dst, int n) {
  int i = blockIdx.x * blockDim.x + threadIdx.x;
  if (i < n) dst[i] = (__bf16)src[i];
}

// ---------------------------------------------------------------------------
// Pack B (KxN, row-major f32) into per-fragment WMMA layout, bf16:
//   dst[((kt*(N/16) + nt)*32 + lane)*16 + e] = B[(kt*32 + 8*(lane>>4) + (e&7) + 16*(e>>3))*N
//                                               + nt*16 + (lane&15)]
// so each lane's 16-element B fragment is one contiguous 32-byte load.
// ---------------------------------------------------------------------------
__global__ void __launch_bounds__(256)
pack_b_f32_to_bf16(const float* __restrict__ src, __bf16* __restrict__ dst,
                   int K, int N) {
  int idx = blockIdx.x * blockDim.x + threadIdx.x;
  if (idx >= K * N) return;
  const int e      = idx & 15;
  const int lane   = (idx >> 4) & 31;
  const int tile   = idx >> 9;
  const int ntiles = N >> 4;
  const int nt     = tile % ntiles;
  const int kt     = tile / ntiles;
  const int n      = lane & 15;
  const int h      = lane >> 4;
  const int k      = (kt << 5) + (h << 3) + (e & 7) + ((e >> 3) << 4);
  dst[idx] = (__bf16)src[k * N + nt * 16 + n];
}

// ---------------------------------------------------------------------------
// bf16 WMMA GEMM: C[M,N] = A[M,K] * Bpacked (+bias per column), f32 out.
// One wave -> 16x64 output tile (4 accumulators, A frag reused 4x).
// A is row-major bf16; B is pre-packed fragment layout (see pack_b).
// Requires M%16==0, N%64==0, K%32==0.
//
// A fragment (ISA 7.12.2): lane (m = lane&15, h = lane>>4) holds
//   elements [k0+8h .. k0+8h+7] and [k0+16+8h .. k0+16+8h+7] of row m
// -> two contiguous 16-byte loads.
// ---------------------------------------------------------------------------
__global__ void __launch_bounds__(128)
gemm_bf16_wmma(const __bf16* __restrict__ A, const __bf16* __restrict__ Bp,
               float* __restrict__ C, const float* __restrict__ bias,
               int M, int N, int K) {
  const int lane    = threadIdx.x & 31;
  const int gw      = (blockIdx.x * blockDim.x + threadIdx.x) >> 5;
  const int ntile64 = N >> 6;
  const int m_tile  = gw / ntile64;
  const int n_tile  = gw % ntile64;
  if (m_tile >= (M >> 4)) return;
  const int row0     = m_tile << 4;
  const int nt0      = n_tile << 2;       // first 16-wide column tile
  const int ntiles16 = N >> 4;
  const int m = lane & 15, h = lane >> 4;

  const __bf16* arow = A + (row0 + m) * K + (h << 3);

  v8f acc0 = {}, acc1 = {}, acc2 = {}, acc3 = {};
  for (int k0 = 0; k0 < K; k0 += 32) {
    // A fragment: two aligned b128 loads, zipped into v16bf
    v8bf lo = *(const v8bf*)(arow + k0);
    v8bf hi = *(const v8bf*)(arow + k0 + 16);
    v16bf a = __builtin_shufflevector(lo, hi, 0, 1, 2, 3, 4, 5, 6, 7,
                                              8, 9, 10, 11, 12, 13, 14, 15);
    __builtin_prefetch(arow + k0 + 32, 0, 0);   // global_prefetch_b8

    // B fragments: one contiguous 32-byte load per lane per 16-col tile
    const __bf16* bbase = Bp + ((((k0 >> 5) * ntiles16) + nt0) << 9) + (lane << 4);
    v16bf b0 = *(const v16bf*)(bbase);
    v16bf b1 = *(const v16bf*)(bbase + 512);
    v16bf b2 = *(const v16bf*)(bbase + 1024);
    v16bf b3 = *(const v16bf*)(bbase + 1536);

    acc0 = __builtin_amdgcn_wmma_f32_16x16x32_bf16(false, a, false, b0, (short)0, acc0, false, false);
    acc1 = __builtin_amdgcn_wmma_f32_16x16x32_bf16(false, a, false, b1, (short)0, acc1, false, false);
    acc2 = __builtin_amdgcn_wmma_f32_16x16x32_bf16(false, a, false, b2, (short)0, acc2, false, false);
    acc3 = __builtin_amdgcn_wmma_f32_16x16x32_bf16(false, a, false, b3, (short)0, acc3, false, false);
  }

  // C/D layout: VGPR r -> M = r + 8*h ; N = lane&15
  const int n = lane & 15;
  v8f accs[4] = {acc0, acc1, acc2, acc3};
#pragma unroll
  for (int j = 0; j < 4; ++j) {
    const int c   = (nt0 + j) * 16 + n;
    const float bv = bias ? bias[c] : 0.0f;
#pragma unroll
    for (int r = 0; r < 8; ++r) {
      C[(row0 + r + 8 * h) * N + c] = accs[j][r] + bv;
    }
  }
}

// ---------------------------------------------------------------------------
// Depthwise causal conv (K=4) + bias + SiLU.  xar is (rows, 2D); cols [0,D).
// ---------------------------------------------------------------------------
__global__ void __launch_bounds__(256)
conv_silu_kernel(const float* __restrict__ xar, const float* __restrict__ conv_w,
                 const float* __restrict__ conv_b, float* __restrict__ xs) {
  int idx = blockIdx.x * blockDim.x + threadIdx.x;
  if (idx >= ROWS * DD) return;
  const int d   = idx % DD;
  const int row = idx / DD;
  const int l   = row % LL;
  const int b   = row / LL;
  float acc = conv_b[d];
#pragma unroll
  for (int k = 0; k < KCONV; ++k) {
    const int lt = l - (KCONV - 1) + k;
    if (lt >= 0)
      acc += conv_w[d * KCONV + k] * xar[(b * LL + lt) * (2 * DD) + d];
  }
  xs[idx] = acc / (1.0f + __expf(-acc));   // SiLU
}

// ---------------------------------------------------------------------------
// x_dbl = xs @ W_x : (2048,1024)x(1024,64).  One thread per output element.
// ---------------------------------------------------------------------------
__global__ void __launch_bounds__(256)
xdbl_kernel(const float* __restrict__ xs, const float* __restrict__ W_x,
            float* __restrict__ xdbl) {
  int idx = blockIdx.x * blockDim.x + threadIdx.x;
  if (idx >= ROWS * (RR + 2 * NSTATE)) return;
  const int c   = idx % (RR + 2 * NSTATE);
  const int row = idx / (RR + 2 * NSTATE);
  const float* xrow = xs + row * DD;
  float acc = 0.0f;
  for (int j = 0; j < DD; ++j)
    acc += xrow[j] * W_x[j * (RR + 2 * NSTATE) + c];
  xdbl[idx] = acc;
}

// ---------------------------------------------------------------------------
// delta = softplus(delta_raw @ W_dt + b_dt) : inner K = 32.
// ---------------------------------------------------------------------------
__global__ void __launch_bounds__(256)
delta_kernel(const float* __restrict__ xdbl, const float* __restrict__ W_dt,
             const float* __restrict__ b_dt, float* __restrict__ delta) {
  int idx = blockIdx.x * blockDim.x + threadIdx.x;
  if (idx >= ROWS * DD) return;
  const int d   = idx % DD;
  const int row = idx / DD;
  const float* xr = xdbl + row * (RR + 2 * NSTATE);
  float acc = b_dt[d];
#pragma unroll
  for (int r = 0; r < RR; ++r)
    acc += xr[r] * W_dt[r * DD + d];
  delta[idx] = (acc > 20.0f) ? acc : log1pf(__expf(acc));
}

// ---------------------------------------------------------------------------
// Selective scan.  One wave32 handles (b, two channels d) x 16 states.
//   s_t = exp(dt*A_n)*s_{t-1} + dt*u*B_n ;  y_t = sum_n s_t*C_n + u*D_skip
// ---------------------------------------------------------------------------
__global__ void __launch_bounds__(256)
scan_kernel(const float* __restrict__ xs, const float* __restrict__ delta,
            const float* __restrict__ xdbl, const float* __restrict__ A_log,
            const float* __restrict__ D_skip, float* __restrict__ y) {
  const int wave = (blockIdx.x * blockDim.x + threadIdx.x) >> 5;
  const int lane = threadIdx.x & 31;
  const int n    = lane & 15;
  const int sub  = lane >> 4;
  const int b    = wave / (DD / 2);
  const int dpr  = wave % (DD / 2);
  const int d    = dpr * 2 + sub;

  const float An = -__expf(A_log[d * NSTATE + n]);
  const float Dd = D_skip[d];
  float s = 0.0f;

  for (int t = 0; t < LL; ++t) {
    const int row = b * LL + t;
    const float dt = delta[row * DD + d];
    const float u  = xs[row * DD + d];
    const float Bn = xdbl[row * (RR + 2 * NSTATE) + RR + n];
    const float Cn = xdbl[row * (RR + 2 * NSTATE) + RR + NSTATE + n];
    s = s * __expf(dt * An) + dt * u * Bn;
    float p = s * Cn;
    p += __shfl_xor(p, 1);
    p += __shfl_xor(p, 2);
    p += __shfl_xor(p, 4);
    p += __shfl_xor(p, 8);
    if (n == 0) y[row * DD + d] = p + u * Dd;
  }
}

// ---------------------------------------------------------------------------
// gated = bf16( y * silu(res) ) ; res = xar[:, D:2D]
// ---------------------------------------------------------------------------
__global__ void __launch_bounds__(256)
gate_kernel(const float* __restrict__ y, const float* __restrict__ xar,
            __bf16* __restrict__ gated) {
  int idx = blockIdx.x * blockDim.x + threadIdx.x;
  if (idx >= ROWS * DD) return;
  const int d   = idx % DD;
  const int row = idx / DD;
  const float res = xar[row * (2 * DD) + DD + d];
  const float g   = res / (1.0f + __expf(-res));
  gated[idx] = (__bf16)(y[idx] * g);
}

// ---------------------------------------------------------------------------
// Host launch
// ---------------------------------------------------------------------------
extern "C" void kernel_launch(void* const* d_in, const int* in_sizes, int n_in,
                              void* d_out, int out_size, void* d_ws, size_t ws_size,
                              hipStream_t stream) {
  const float* x      = (const float*)d_in[0];   // (B,L,DIM)
  const float* W_in   = (const float*)d_in[1];   // (DIM,2D)
  const float* conv_w = (const float*)d_in[2];   // (D,1,K)
  const float* conv_b = (const float*)d_in[3];   // (D,)
  const float* W_x    = (const float*)d_in[4];   // (D,R+2N)
  const float* W_dt   = (const float*)d_in[5];   // (R,D)
  const float* b_dt   = (const float*)d_in[6];   // (D,)
  const float* A_log  = (const float*)d_in[7];   // (D,N)
  const float* D_skip = (const float*)d_in[8];   // (D,)
  const float* W_out  = (const float*)d_in[9];   // (D,DIM)
  const float* b_out  = (const float*)d_in[10];  // (DIM,)
  float* out = (float*)d_out;                    // (B,L,DIM)

  // workspace layout (256B aligned chunks)
  char* ws = (char*)d_ws;
  size_t off = 0;
  auto carve = [&](size_t bytes) {
    void* p = ws + off;
    off += (bytes + 255) & ~(size_t)255;
    return p;
  };
  __bf16* x_bf   = (__bf16*)carve((size_t)ROWS * DIMM * 2);       // 2 MB
  __bf16* Wi_pk  = (__bf16*)carve((size_t)DIMM * 2 * DD * 2);     // 2 MB
  __bf16* Wo_pk  = (__bf16*)carve((size_t)DD * DIMM * 2);         // 1 MB
  float*  xar    = (float*) carve((size_t)ROWS * 2 * DD * 4);     // 16 MB
  float*  xs     = (float*) carve((size_t)ROWS * DD * 4);         // 8 MB
  float*  xdbl   = (float*) carve((size_t)ROWS * (RR + 2*NSTATE) * 4);
  float*  delta  = (float*) carve((size_t)ROWS * DD * 4);         // 8 MB
  float*  yv     = (float*) carve((size_t)ROWS * DD * 4);         // 8 MB
  __bf16* gated  = (__bf16*)carve((size_t)ROWS * DD * 2);         // 4 MB

  // 1) stage bf16 operands: A row-major, B fragment-packed
  {
    int n = ROWS * DIMM;
    cvt_f32_to_bf16<<<(n + 255) / 256, 256, 0, stream>>>(x, x_bf, n);
  }
  {
    int n = DIMM * 2 * DD;
    pack_b_f32_to_bf16<<<(n + 255) / 256, 256, 0, stream>>>(W_in, Wi_pk,
                                                            DIMM, 2 * DD);
  }
  {
    int n = DD * DIMM;
    pack_b_f32_to_bf16<<<(n + 255) / 256, 256, 0, stream>>>(W_out, Wo_pk,
                                                            DD, DIMM);
  }

  // 2) x_and_res = x @ W_in   (M=2048, N=2048, K=512)
  {
    int waves = (ROWS / 16) * ((2 * DD) / 64);     // 4096
    gemm_bf16_wmma<<<waves / 4, 128, 0, stream>>>(x_bf, Wi_pk, xar, nullptr,
                                                  ROWS, 2 * DD, DIMM);
  }

  // 3) depthwise causal conv + bias + SiLU
  conv_silu_kernel<<<(ROWS * DD) / 256, 256, 0, stream>>>(xar, conv_w, conv_b, xs);

  // 4) x_dbl = xs @ W_x
  {
    int n = ROWS * (RR + 2 * NSTATE);
    xdbl_kernel<<<(n + 255) / 256, 256, 0, stream>>>(xs, W_x, xdbl);
  }

  // 5) delta = softplus(delta_raw @ W_dt + b_dt)
  delta_kernel<<<(ROWS * DD) / 256, 256, 0, stream>>>(xdbl, W_dt, b_dt, delta);

  // 6) selective scan
  {
    int waves = BB * (DD / 2);   // 1024 waves
    scan_kernel<<<waves * 32 / 256, 256, 0, stream>>>(xs, delta, xdbl, A_log,
                                                      D_skip, yv);
  }

  // 7) gate: y * silu(res) -> bf16
  gate_kernel<<<(ROWS * DD) / 256, 256, 0, stream>>>(yv, xar, gated);

  // 8) out = gated @ W_out + b_out  (M=2048, N=512, K=1024)
  {
    int waves = (ROWS / 16) * (DIMM / 64);         // 1024
    gemm_bf16_wmma<<<waves / 4, 128, 0, stream>>>(gated, Wo_pk, out, b_out,
                                                  ROWS, DIMM, DD);
  }
}
